// DeepseekV2DecoderLayer_4664334484019
// MI455X (gfx1250) — compile-verified, matrix operand order checked by reading, not run on hardware
//
#include <hip/hip_runtime.h>
#include <hip/hip_bf16.h>

// ---------------------------------------------------------------------------
// DeepseekV2 decoder layer for MI455X (gfx1250, wave32, WMMA bf16 16x16x32,
// async global->LDS staging on the ASYNCcnt path where available).
// ---------------------------------------------------------------------------

#define S_LEN 2048
#define HDIM  2048
#define NHEAD 16
#define NOPE_D 128
#define ROPE_HALF 32
#define KVL_D 512
#define VH_D  128
#define QHD_D 192
#define DQK   576      // KVL + ROPE
#define IDIM  10944

typedef __bf16 bf16_t;
typedef bf16_t v16bf __attribute__((ext_vector_type(16)));
typedef float  v8f   __attribute__((ext_vector_type(8)));
typedef int    v4i_t __attribute__((ext_vector_type(4)));

#define V8ZERO {0.f,0.f,0.f,0.f,0.f,0.f,0.f,0.f}

#if __has_builtin(__builtin_amdgcn_global_load_async_to_lds_b128) && \
    __has_builtin(__builtin_amdgcn_s_wait_asynccnt)
#define HAVE_ASYNC_LDS 1
#else
#define HAVE_ASYNC_LDS 0
#endif

__device__ __forceinline__ v8f wmma_bf16(v16bf a, v16bf b, v8f c) {
  // D = A(16x32 bf16) * B(32x16 bf16) + C(16x16 f32)
  return __builtin_amdgcn_wmma_f32_16x16x32_bf16(
      false, a, false, b, (short)0, c, false, false);
}

__device__ __forceinline__ unsigned pack_bf16(float a, float b) {
  union { bf16_t h[2]; unsigned u; } p;
  p.h[0] = (bf16_t)a; p.h[1] = (bf16_t)b;
  return p.u;
}

// 16B global -> LDS copy; async (ASYNCcnt-tracked, no VGPR round trip) when
// the gfx1250 builtin is available, else synchronous uint4 copy.
// Probe-derived prototype: (int4* global_src, int4* lds_dst, Imm off, Imm cpol)
__device__ __forceinline__ void copy_b128_to_lds(const bf16_t* g, bf16_t* l) {
#if HAVE_ASYNC_LDS
  __builtin_amdgcn_global_load_async_to_lds_b128((v4i_t*)g, (v4i_t*)l, 0, 0);
#else
  *(uint4*)l = *(const uint4*)g;
#endif
}
__device__ __forceinline__ void async_join() {
#if HAVE_ASYNC_LDS
  __builtin_amdgcn_s_wait_asynccnt(0);
#endif
}

// Load a 16x32 bf16 fragment (A layout, or B when tile is stored [n][k]).
// ISA 7.12.2: lanes 0-15 hold row = lane, K pairs {0..7,16..23}; lanes 16-31
// hold K pairs {8..15,24..31}. Pairs are consecutive-K => 32-bit loads
// (compiler merges into ds_load_b128).
__device__ __forceinline__ v16bf load_frag(const bf16_t* base, int ldRow, int lane) {
  union { v16bf v; unsigned u[8]; } f;
  const bf16_t* p = base + (size_t)(lane & 15) * (size_t)ldRow;
  const int kb = (lane >> 4) << 3;
#pragma unroll
  for (int i = 0; i < 4; ++i) {
    f.u[i]     = *(const unsigned*)(p + kb + 2 * i);
    f.u[i + 4] = *(const unsigned*)(p + 16 + kb + 2 * i);
  }
  return f.v;
}

// ---------------------------------------------------------------------------
// RMSNorm (f32 in, bf16 out): one row per workgroup.
// ---------------------------------------------------------------------------
__global__ __launch_bounds__(256) void rmsnorm_bf16_kernel(
    const float* __restrict__ x, const float* __restrict__ w,
    bf16_t* __restrict__ out, int H) {
  const int s = blockIdx.x;
  const float* row = x + (size_t)s * H;
  __shared__ float red[256];
  float ss = 0.f;
  for (int i = threadIdx.x; i < H; i += 256) { float v = row[i]; ss += v * v; }
  red[threadIdx.x] = ss;
  __syncthreads();
  for (int o = 128; o > 0; o >>= 1) {
    if (threadIdx.x < (unsigned)o) red[threadIdx.x] += red[threadIdx.x + o];
    __syncthreads();
  }
  const float inv = rsqrtf(red[0] / (float)H + 1e-6f);
  for (int i = threadIdx.x; i < H; i += 256)
    out[(size_t)s * H + i] = (bf16_t)(row[i] * inv * w[i]);
}

// ---------------------------------------------------------------------------
// Generic tiled GEMM: C[M,N] = A_bf16[M,K] * B_f32->bf16[K,N]
//   Block tile 128x64, 8 waves (4x2), 32x32 per wave -> 4 WMMAs/k-step.
//   Double-buffered LDS pipeline: async A copy + early B register loads for
//   tile i+1 issued before computing tile i; one barrier per k-step.
//   OUT_BF16: store bf16; RES: add f32 residual. Batched via blockIdx.z.
// ---------------------------------------------------------------------------
template <int OUT_BF16, int RES>
__global__ __launch_bounds__(256) void gemm_kernel(
    const bf16_t* __restrict__ A, int lda, long long sA,
    const float* __restrict__ B, int ldb, long long sB,
    void* __restrict__ Cv, int ldc, long long sC,
    const float* __restrict__ Res, int ldres,
    int K) {
  __shared__ __align__(16) bf16_t As[2][128][32];
  __shared__ __align__(16) bf16_t Bs[2][64][32];   // [n][k] (transposed)
  const int bn = blockIdx.x * 64;
  const int bm = blockIdx.y * 128;
  const int z  = blockIdx.z;
  A += (size_t)z * (size_t)sA;
  B += (size_t)z * (size_t)sB;
  const int tid = threadIdx.x, lane = tid & 31, wave = tid >> 5;
  const int wm = (wave & 3) * 32;
  const int wn = (wave >> 2) * 32;
  const int ar = tid >> 1, acb = (tid & 1) << 4;     // A staging map
  const int kp = tid & 15, bcb = (tid >> 4) << 2;    // B staging map (k-pairs)
  const bf16_t* Arow = A + (size_t)(bm + ar) * lda + acb;
  const float*  Brow = B + (size_t)(2 * kp) * ldb + bn + bcb;

  v8f acc00 = V8ZERO, acc01 = V8ZERO, acc10 = V8ZERO, acc11 = V8ZERO;

  // ---- prologue: stage tile 0 ----
  copy_b128_to_lds(Arow,     &As[0][ar][acb]);
  copy_b128_to_lds(Arow + 8, &As[0][ar][acb + 8]);
  float4 f0 = *(const float4*)(Brow);
  float4 f1 = *(const float4*)(Brow + ldb);
  async_join();
  *(unsigned*)(&Bs[0][bcb + 0][2 * kp]) = pack_bf16(f0.x, f1.x);
  *(unsigned*)(&Bs[0][bcb + 1][2 * kp]) = pack_bf16(f0.y, f1.y);
  *(unsigned*)(&Bs[0][bcb + 2][2 * kp]) = pack_bf16(f0.z, f1.z);
  *(unsigned*)(&Bs[0][bcb + 3][2 * kp]) = pack_bf16(f0.w, f1.w);
  __syncthreads();

  for (int k0 = 0; k0 < K; k0 += 32) {
    const int cur = (k0 >> 5) & 1, nxt = cur ^ 1;
    const bool more = (k0 + 32) < K;
    if (more) {   // issue next tile's loads before computing current tile
      const bf16_t* an = Arow + k0 + 32;
      copy_b128_to_lds(an,     &As[nxt][ar][acb]);
      copy_b128_to_lds(an + 8, &As[nxt][ar][acb + 8]);
      const float* bnx = Brow + (size_t)(k0 + 32) * ldb;
      f0 = *(const float4*)(bnx);
      f1 = *(const float4*)(bnx + ldb);
    }
    v16bf a0 = load_frag(&As[cur][wm][0],      32, lane);
    v16bf a1 = load_frag(&As[cur][wm + 16][0], 32, lane);
    v16bf b0 = load_frag(&Bs[cur][wn][0],      32, lane);
    v16bf b1 = load_frag(&Bs[cur][wn + 16][0], 32, lane);
    acc00 = wmma_bf16(a0, b0, acc00);
    acc01 = wmma_bf16(a0, b1, acc01);
    acc10 = wmma_bf16(a1, b0, acc10);
    acc11 = wmma_bf16(a1, b1, acc11);
    if (more) {
      async_join();
      *(unsigned*)(&Bs[nxt][bcb + 0][2 * kp]) = pack_bf16(f0.x, f1.x);
      *(unsigned*)(&Bs[nxt][bcb + 1][2 * kp]) = pack_bf16(f0.y, f1.y);
      *(unsigned*)(&Bs[nxt][bcb + 2][2 * kp]) = pack_bf16(f0.z, f1.z);
      *(unsigned*)(&Bs[nxt][bcb + 3][2 * kp]) = pack_bf16(f0.w, f1.w);
    }
    __syncthreads();
  }

  const int lh  = (lane >> 4) << 3;
  const int col = bn + wn + (lane & 15);
#pragma unroll
  for (int r = 0; r < 8; ++r) {
    const int m0 = bm + wm + lh + r;
    const int m1 = m0 + 16;
    float v00 = acc00[r], v01 = acc01[r];
    float v10 = acc10[r], v11 = acc11[r];
    if (RES) {
      v00 += Res[(size_t)m0 * ldres + col];
      v01 += Res[(size_t)m0 * ldres + col + 16];
      v10 += Res[(size_t)m1 * ldres + col];
      v11 += Res[(size_t)m1 * ldres + col + 16];
    }
    if (OUT_BF16) {
      bf16_t* C = (bf16_t*)Cv + (size_t)z * (size_t)sC;
      C[(size_t)m0 * ldc + col]      = (bf16_t)v00;
      C[(size_t)m0 * ldc + col + 16] = (bf16_t)v01;
      C[(size_t)m1 * ldc + col]      = (bf16_t)v10;
      C[(size_t)m1 * ldc + col + 16] = (bf16_t)v11;
    } else {
      float* C = (float*)Cv + (size_t)z * (size_t)sC;
      C[(size_t)m0 * ldc + col]      = v00;
      C[(size_t)m0 * ldc + col + 16] = v01;
      C[(size_t)m1 * ldc + col]      = v10;
      C[(size_t)m1 * ldc + col + 16] = v11;
    }
  }
}

// ---------------------------------------------------------------------------
// Fused gate_up GEMM + SiLU: act = silu(x2 @ Wg) * (x2 @ Wu), bf16 out.
// Same double-buffered pipeline; 8 WMMAs per k-step per wave; v_rcp SiLU.
// ---------------------------------------------------------------------------
__global__ __launch_bounds__(256) void gateup_kernel(
    const bf16_t* __restrict__ A, int lda,
    const float* __restrict__ W,       // [HDIM][2*IDIM]
    bf16_t* __restrict__ act) {
  __shared__ __align__(16) bf16_t As[2][128][32];
  __shared__ __align__(16) bf16_t Gs[2][64][32];
  __shared__ __align__(16) bf16_t Us[2][64][32];
  const int bn = blockIdx.x * 64, bm = blockIdx.y * 128;
  const int tid = threadIdx.x, lane = tid & 31, wave = tid >> 5;
  const int wm = (wave & 3) * 32, wn = (wave >> 2) * 32;
  const int ldb = 2 * IDIM;
  const int ar = tid >> 1, acb = (tid & 1) << 4;
  const int kp = tid & 15, bcb = (tid >> 4) << 2;
  const bf16_t* Arow = A + (size_t)(bm + ar) * lda + acb;
  const float*  Wrow = W + (size_t)(2 * kp) * ldb + bn + bcb;

  v8f g00 = V8ZERO, g01 = V8ZERO, g10 = V8ZERO, g11 = V8ZERO;
  v8f u00 = V8ZERO, u01 = V8ZERO, u10 = V8ZERO, u11 = V8ZERO;

  // ---- prologue ----
  copy_b128_to_lds(Arow,     &As[0][ar][acb]);
  copy_b128_to_lds(Arow + 8, &As[0][ar][acb + 8]);
  float4 fg0 = *(const float4*)(Wrow);
  float4 fg1 = *(const float4*)(Wrow + ldb);
  float4 fu0 = *(const float4*)(Wrow + IDIM);
  float4 fu1 = *(const float4*)(Wrow + IDIM + ldb);
  async_join();
  *(unsigned*)(&Gs[0][bcb + 0][2 * kp]) = pack_bf16(fg0.x, fg1.x);
  *(unsigned*)(&Gs[0][bcb + 1][2 * kp]) = pack_bf16(fg0.y, fg1.y);
  *(unsigned*)(&Gs[0][bcb + 2][2 * kp]) = pack_bf16(fg0.z, fg1.z);
  *(unsigned*)(&Gs[0][bcb + 3][2 * kp]) = pack_bf16(fg0.w, fg1.w);
  *(unsigned*)(&Us[0][bcb + 0][2 * kp]) = pack_bf16(fu0.x, fu1.x);
  *(unsigned*)(&Us[0][bcb + 1][2 * kp]) = pack_bf16(fu0.y, fu1.y);
  *(unsigned*)(&Us[0][bcb + 2][2 * kp]) = pack_bf16(fu0.z, fu1.z);
  *(unsigned*)(&Us[0][bcb + 3][2 * kp]) = pack_bf16(fu0.w, fu1.w);
  __syncthreads();

  for (int k0 = 0; k0 < HDIM; k0 += 32) {
    const int cur = (k0 >> 5) & 1, nxt = cur ^ 1;
    const bool more = (k0 + 32) < HDIM;
    if (more) {
      const bf16_t* an = Arow + k0 + 32;
      copy_b128_to_lds(an,     &As[nxt][ar][acb]);
      copy_b128_to_lds(an + 8, &As[nxt][ar][acb + 8]);
      const float* wn2 = Wrow + (size_t)(k0 + 32) * ldb;
      fg0 = *(const float4*)(wn2);
      fg1 = *(const float4*)(wn2 + ldb);
      fu0 = *(const float4*)(wn2 + IDIM);
      fu1 = *(const float4*)(wn2 + IDIM + ldb);
    }
    v16bf a0  = load_frag(&As[cur][wm][0],      32, lane);
    v16bf a1  = load_frag(&As[cur][wm + 16][0], 32, lane);
    v16bf bg0 = load_frag(&Gs[cur][wn][0],      32, lane);
    v16bf bg1 = load_frag(&Gs[cur][wn + 16][0], 32, lane);
    v16bf bu0 = load_frag(&Us[cur][wn][0],      32, lane);
    v16bf bu1 = load_frag(&Us[cur][wn + 16][0], 32, lane);
    g00 = wmma_bf16(a0, bg0, g00);
    g01 = wmma_bf16(a0, bg1, g01);
    g10 = wmma_bf16(a1, bg0, g10);
    g11 = wmma_bf16(a1, bg1, g11);
    u00 = wmma_bf16(a0, bu0, u00);
    u01 = wmma_bf16(a0, bu1, u01);
    u10 = wmma_bf16(a1, bu0, u10);
    u11 = wmma_bf16(a1, bu1, u11);
    if (more) {
      async_join();
      *(unsigned*)(&Gs[nxt][bcb + 0][2 * kp]) = pack_bf16(fg0.x, fg1.x);
      *(unsigned*)(&Gs[nxt][bcb + 1][2 * kp]) = pack_bf16(fg0.y, fg1.y);
      *(unsigned*)(&Gs[nxt][bcb + 2][2 * kp]) = pack_bf16(fg0.z, fg1.z);
      *(unsigned*)(&Gs[nxt][bcb + 3][2 * kp]) = pack_bf16(fg0.w, fg1.w);
      *(unsigned*)(&Us[nxt][bcb + 0][2 * kp]) = pack_bf16(fu0.x, fu1.x);
      *(unsigned*)(&Us[nxt][bcb + 1][2 * kp]) = pack_bf16(fu0.y, fu1.y);
      *(unsigned*)(&Us[nxt][bcb + 2][2 * kp]) = pack_bf16(fu0.z, fu1.z);
      *(unsigned*)(&Us[nxt][bcb + 3][2 * kp]) = pack_bf16(fu0.w, fu1.w);
    }
    __syncthreads();
  }

  const int lh  = (lane >> 4) << 3;
  const int col = bn + wn + (lane & 15);
#pragma unroll
  for (int r = 0; r < 8; ++r) {
    const int m0 = bm + wm + lh + r;
    const int m1 = m0 + 16;
    float g, u;
    g = g00[r]; u = u00[r];
    act[(size_t)m0 * IDIM + col] =
        (bf16_t)(g * __builtin_amdgcn_rcpf(1.f + __expf(-g)) * u);
    g = g01[r]; u = u01[r];
    act[(size_t)m0 * IDIM + col + 16] =
        (bf16_t)(g * __builtin_amdgcn_rcpf(1.f + __expf(-g)) * u);
    g = g10[r]; u = u10[r];
    act[(size_t)m1 * IDIM + col] =
        (bf16_t)(g * __builtin_amdgcn_rcpf(1.f + __expf(-g)) * u);
    g = g11[r]; u = u11[r];
    act[(size_t)m1 * IDIM + col + 16] =
        (bf16_t)(g * __builtin_amdgcn_rcpf(1.f + __expf(-g)) * u);
  }
}

// ---------------------------------------------------------------------------
// q prep: split q into q_nope (bf16, [s][h*128+d]) and roped q_pe written
// into query[s][h][512..576) bf16.
// ---------------------------------------------------------------------------
__global__ __launch_bounds__(256) void prep_q_kernel(
    const float* __restrict__ q, const float* __restrict__ cosb,
    const float* __restrict__ sinb, bf16_t* __restrict__ qn,
    bf16_t* __restrict__ query) {
  const int s = blockIdx.x, t = threadIdx.x;
  const float* qs = q + (size_t)s * (NHEAD * QHD_D);
  for (int i = t; i < NHEAD * NOPE_D; i += 256) {
    const int h = i >> 7, d = i & 127;
    qn[(size_t)s * (NHEAD * NOPE_D) + i] = (bf16_t)qs[h * QHD_D + d];
  }
  for (int i = t; i < NHEAD * 64; i += 256) {
    const int h = i >> 6, d = i & 63;
    const float v = qs[h * QHD_D + NOPE_D + d];
    const float rot = (d < ROPE_HALF) ? -qs[h * QHD_D + NOPE_D + d + ROPE_HALF]
                                      :  qs[h * QHD_D + NOPE_D + d - ROPE_HALF];
    const float o = v * cosb[s * 64 + d] + rot * sinb[s * 64 + d];
    query[(size_t)s * (NHEAD * DQK) + h * DQK + KVL_D + d] = (bf16_t)o;
  }
}

// ---------------------------------------------------------------------------
// kv prep: key[s][0..512) = rms(kv[:512])*w (== v_c), key[s][512..576) = rope(k_pe)
// ---------------------------------------------------------------------------
__global__ __launch_bounds__(256) void prep_kv_kernel(
    const float* __restrict__ kv, const float* __restrict__ lnw,
    const float* __restrict__ cosb, const float* __restrict__ sinb,
    bf16_t* __restrict__ key) {
  const int s = blockIdx.x, t = threadIdx.x;
  const float* row = kv + (size_t)s * DQK;
  __shared__ float red[256];
  float ss = 0.f;
  for (int i = t; i < KVL_D; i += 256) { float v = row[i]; ss += v * v; }
  red[t] = ss;
  __syncthreads();
  for (int o = 128; o > 0; o >>= 1) {
    if (t < o) red[t] += red[t + o];
    __syncthreads();
  }
  const float inv = rsqrtf(red[0] / (float)KVL_D + 1e-6f);
  for (int i = t; i < KVL_D; i += 256)
    key[(size_t)s * DQK + i] = (bf16_t)(row[i] * inv * lnw[i]);
  if (t < 64) {
    const int d = t;
    const float v = row[KVL_D + d];
    const float rot = (d < ROPE_HALF) ? -row[KVL_D + d + ROPE_HALF]
                                      :  row[KVL_D + d - ROPE_HALF];
    key[(size_t)s * DQK + KVL_D + d] =
        (bf16_t)(v * cosb[s * 64 + d] + rot * sinb[s * 64 + d]);
  }
}

// ---------------------------------------------------------------------------
// Attention pass 1: per-row running max/sum-exp (flash stats), causal.
// Block: 64 q-rows x one head. 8 waves = 4 q-subtiles x 2 k-subtiles.
// Scores computed with WMMA directly from global (192MB L2 backs reuse).
// ---------------------------------------------------------------------------
__global__ __launch_bounds__(256) void attn_stats_kernel(
    const bf16_t* __restrict__ query, const bf16_t* __restrict__ key,
    float* __restrict__ statm, float* __restrict__ statl) {
  const int q0 = blockIdx.x * 64;
  const int h  = blockIdx.y;
  const int tid = threadIdx.x, lane = tid & 31, wave = tid >> 5;
  const int qsub = wave & 3, ksub = wave >> 2;
  __shared__ float rm[64], rl[64], pm[2][64], pl[2][64];
  if (tid < 64) { rm[tid] = -1e30f; rl[tid] = 0.f; }
  __syncthreads();
  const float scale = 0.07216878364870322f;   // 192^-0.5
  const bf16_t* qbase =
      query + (size_t)(q0 + qsub * 16) * (NHEAD * DQK) + h * DQK;
  const int nk = (q0 + 64) / 32;
  for (int kt = 0; kt < nk; ++kt) {
    const int k0 = kt * 32 + ksub * 16;
    const bf16_t* kbase = key + (size_t)k0 * DQK;
    v8f acc = V8ZERO;
    for (int d0 = 0; d0 < DQK; d0 += 32) {
      v16bf a = load_frag(qbase + d0, NHEAD * DQK, lane);
      v16bf b = load_frag(kbase + d0, DQK, lane);
      acc = wmma_bf16(a, b, acc);
    }
    const int mb = q0 + qsub * 16 + ((lane >> 4) << 3);
    const int n  = k0 + (lane & 15);
    float sv[8];
#pragma unroll
    for (int r = 0; r < 8; ++r)
      sv[r] = (n > mb + r) ? -1e30f : acc[r] * scale;
    float mrow[8], srow[8];
#pragma unroll
    for (int r = 0; r < 8; ++r) {   // 16-lane-half row reductions (wave32)
      float v = sv[r];
      v = fmaxf(v, __shfl_xor(v, 1, 32));
      v = fmaxf(v, __shfl_xor(v, 2, 32));
      v = fmaxf(v, __shfl_xor(v, 4, 32));
      v = fmaxf(v, __shfl_xor(v, 8, 32));
      mrow[r] = v;
      float e = __expf(sv[r] - v);
      e += __shfl_xor(e, 1, 32);
      e += __shfl_xor(e, 2, 32);
      e += __shfl_xor(e, 4, 32);
      e += __shfl_xor(e, 8, 32);
      srow[r] = e;
    }
    if ((lane & 15) == 0) {
      const int rb = qsub * 16 + ((lane >> 4) << 3);
#pragma unroll
      for (int r = 0; r < 8; ++r) {
        pm[ksub][rb + r] = mrow[r];
        pl[ksub][rb + r] = srow[r];
      }
    }
    __syncthreads();
    if (tid < 64) {   // merge both k-halves into running stats
#pragma unroll
      for (int kh = 0; kh < 2; ++kh) {
        const float m2 = pm[kh][tid], l2 = pl[kh][tid];
        const float mn = fmaxf(rm[tid], m2);
        rl[tid] = rl[tid] * __expf(rm[tid] - mn) + l2 * __expf(m2 - mn);
        rm[tid] = mn;
      }
    }
    __syncthreads();
  }
  if (tid < 64) {
    statm[(size_t)h * S_LEN + q0 + tid] = rm[tid];
    statl[(size_t)h * S_LEN + q0 + tid] = rl[tid];
  }
}

// ---------------------------------------------------------------------------
// Attention pass 2: attn[q,h,:512] = softmax(scores) @ v_c using saved stats.
// Block: 32 q-rows x one head. 8 waves = 2 q-halves x 4 v-col chunks (128).
// Waves 0-1 recompute the 32x32 score tile -> P~ = exp(s-m) in LDS; all
// waves run P~ @ V with WMMA; scale by precomputed 1/l at the end.
// ---------------------------------------------------------------------------
__global__ __launch_bounds__(256) void attn_pv_kernel(
    const bf16_t* __restrict__ query, const bf16_t* __restrict__ key,
    const float* __restrict__ statm, const float* __restrict__ statl,
    bf16_t* __restrict__ attn) {
  const int q0 = blockIdx.x * 32;
  const int h  = blockIdx.y;
  const int tid = threadIdx.x, lane = tid & 31, wave = tid >> 5;
  const int wr = wave & 1, wc = wave >> 1;
  __shared__ __align__(16) bf16_t plds[32][32];
  __shared__ __align__(16) bf16_t vtile[512][32];   // [col][k]
  __shared__ float mrow[32], linv[32];
  if (tid < 32) {
    mrow[tid] = statm[(size_t)h * S_LEN + q0 + tid];
    linv[tid] = __builtin_amdgcn_rcpf(statl[(size_t)h * S_LEN + q0 + tid]);
  }
  __syncthreads();
  const float scale = 0.07216878364870322f;
  v8f o[8];
#pragma unroll
  for (int f = 0; f < 8; ++f) { v8f z = V8ZERO; o[f] = z; }
  const int nk = q0 / 32 + 1;
  for (int kt = 0; kt < nk; ++kt) {
    const int k0 = kt * 32;
    {   // stage V transposed, bf16 pairs packed along k
      const int kp = tid & 15;
      const int cb = (tid >> 4) * 32;
      const bf16_t* s0 = key + (size_t)(k0 + 2 * kp) * DQK + cb;
      const bf16_t* s1 = s0 + DQK;
#pragma unroll 8
      for (int j = 0; j < 32; ++j) {
        union { bf16_t h2[2]; unsigned u; } p;
        p.h2[0] = s0[j]; p.h2[1] = s1[j];
        *(unsigned*)(&vtile[cb + j][2 * kp]) = p.u;
      }
    }
    if (wave < 2) {   // recompute 16x32 score rows, write P~ to LDS
      const bf16_t* qb =
          query + (size_t)(q0 + wave * 16) * (NHEAD * DQK) + h * DQK;
      const bf16_t* kb = key + (size_t)k0 * DQK;
      v8f s0 = V8ZERO, s1 = V8ZERO;
      for (int d0 = 0; d0 < DQK; d0 += 32) {
        v16bf a  = load_frag(qb + d0, NHEAD * DQK, lane);
        v16bf b0 = load_frag(kb + d0, DQK, lane);
        v16bf b1 = load_frag(kb + (size_t)16 * DQK + d0, DQK, lane);
        s0 = wmma_bf16(a, b0, s0);
        s1 = wmma_bf16(a, b1, s1);
      }
      const int rb = wave * 16 + ((lane >> 4) << 3);
      const int n  = lane & 15;
#pragma unroll
      for (int r = 0; r < 8; ++r) {
        const int row = rb + r;
        const int gq  = q0 + row;
        const float p0 =
            (k0 + n > gq) ? 0.f : __expf(s0[r] * scale - mrow[row]);
        const float p1 =
            (k0 + 16 + n > gq) ? 0.f : __expf(s1[r] * scale - mrow[row]);
        plds[row][n]      = (bf16_t)p0;
        plds[row][16 + n] = (bf16_t)p1;
      }
    }
    __syncthreads();
    {
      v16bf a = load_frag(&plds[wr * 16][0], 32, lane);
#pragma unroll
      for (int f = 0; f < 8; ++f) {
        v16bf b = load_frag(&vtile[wc * 128 + f * 16][0], 32, lane);
        o[f] = wmma_bf16(a, b, o[f]);
      }
    }
    __syncthreads();
  }
  const int rb = wr * 16 + ((lane >> 4) << 3);
  const int n  = lane & 15;
#pragma unroll
  for (int f = 0; f < 8; ++f) {
#pragma unroll
    for (int r = 0; r < 8; ++r) {
      const int row = rb + r;
      const float v = o[f][r] * linv[row];
      attn[(size_t)(q0 + row) * (NHEAD * KVL_D) + h * KVL_D +
           wc * 128 + f * 16 + n] = (bf16_t)v;
    }
  }
}

// ---------------------------------------------------------------------------
// Host-side orchestration.
// ---------------------------------------------------------------------------
extern "C" void kernel_launch(void* const* d_in, const int* in_sizes, int n_in,
                              void* d_out, int out_size, void* d_ws,
                              size_t ws_size, hipStream_t stream) {
  (void)in_sizes; (void)n_in; (void)out_size; (void)ws_size;
  const float* hidden    = (const float*)d_in[0];
  const float* cosb      = (const float*)d_in[1];
  const float* sinb      = (const float*)d_in[2];
  const float* ln1_w     = (const float*)d_in[3];
  const float* q_proj_w  = (const float*)d_in[4];
  const float* kv_a_w    = (const float*)d_in[5];
  const float* kv_a_ln_w = (const float*)d_in[6];
  const float* kc_w      = (const float*)d_in[7];
  const float* vc_w      = (const float*)d_in[8];
  const float* o_proj_w  = (const float*)d_in[9];
  const float* ln2_w     = (const float*)d_in[10];
  const float* gate_up_w = (const float*)d_in[11];
  const float* down_w    = (const float*)d_in[12];
  float* out = (float*)d_out;

  char* ws = (char*)d_ws;
  size_t off = 0;
  auto take = [&](size_t bytes) -> char* {
    char* p = ws + off;
    off += (bytes + 255) & ~(size_t)255;
    return p;
  };
  const size_t S = S_LEN;
  bf16_t* xb    = (bf16_t*)take(S * HDIM * 2);                 //  8.4 MB
  float*  qf    = (float*) take(S * NHEAD * QHD_D * 4);        // 25.2 MB
  float*  kvf   = (float*) take(S * DQK * 4);                  //  4.7 MB
  bf16_t* qn    = (bf16_t*)take(S * NHEAD * NOPE_D * 2);       //  8.4 MB
  bf16_t* query = (bf16_t*)take(S * NHEAD * DQK * 2);          // 37.7 MB
  bf16_t* key   = (bf16_t*)take(S * DQK * 2);                  //  2.4 MB
  float*  statm = (float*) take((size_t)NHEAD * S * 4);
  float*  statl = (float*) take((size_t)NHEAD * S * 4);
  bf16_t* attn  = (bf16_t*)take(S * NHEAD * KVL_D * 2);        // 33.6 MB
  bf16_t* ob    = (bf16_t*)take(S * HDIM * 2);                 //  8.4 MB
  float*  hbuf  = (float*) take(S * HDIM * 4);                 // 16.8 MB
  bf16_t* x2b   = (bf16_t*)take(S * HDIM * 2);                 //  8.4 MB
  bf16_t* act   = (bf16_t*)take(S * IDIM * 2);                 // 44.8 MB

  // 1) x = rms(hidden, ln1_w) -> bf16
  rmsnorm_bf16_kernel<<<S_LEN, 256, 0, stream>>>(hidden, ln1_w, xb, HDIM);
  // 2) q = x @ q_proj_w  (S x 3072, f32)
  gemm_kernel<0, 0><<<dim3(3072 / 64, S_LEN / 128, 1), 256, 0, stream>>>(
      xb, HDIM, 0, q_proj_w, NHEAD * QHD_D, 0, qf, NHEAD * QHD_D, 0,
      nullptr, 0, HDIM);
  // 3) kv = x @ kv_a_w  (S x 576, f32)
  gemm_kernel<0, 0><<<dim3(DQK / 64, S_LEN / 128, 1), 256, 0, stream>>>(
      xb, HDIM, 0, kv_a_w, DQK, 0, kvf, DQK, 0, nullptr, 0, HDIM);
  // 4) split q -> qn bf16 + rope(q_pe) into query[...,512:576)
  prep_q_kernel<<<S_LEN, 256, 0, stream>>>(qf, cosb, sinb, qn, query);
  // 5) key = [rms(kv[:512]) | rope(k_pe)]  bf16
  prep_kv_kernel<<<S_LEN, 256, 0, stream>>>(kvf, kv_a_ln_w, cosb, sinb, key);
  // 6) query[...,0:512) = q_nope @ kc_w  (batched over heads, bf16 out)
  gemm_kernel<1, 0><<<dim3(KVL_D / 64, S_LEN / 128, NHEAD), 256, 0, stream>>>(
      qn, NHEAD * NOPE_D, NOPE_D, kc_w, KVL_D, (long long)NOPE_D * KVL_D,
      query, NHEAD * DQK, DQK, nullptr, 0, NOPE_D);
  // 7-8) two-pass flash attention
  attn_stats_kernel<<<dim3(S_LEN / 64, NHEAD), 256, 0, stream>>>(
      query, key, statm, statl);
  attn_pv_kernel<<<dim3(S_LEN / 32, NHEAD), 256, 0, stream>>>(
      query, key, statm, statl, attn);
  // 9) o = attn @ vc_w  (batched over heads, bf16 out, packed S x 2048)
  gemm_kernel<1, 0><<<dim3(VH_D / 64, S_LEN / 128, NHEAD), 256, 0, stream>>>(
      attn, NHEAD * KVL_D, KVL_D, vc_w, VH_D, (long long)KVL_D * VH_D,
      ob, HDIM, VH_D, nullptr, 0, KVL_D);
  // 10) h = hidden + o @ o_proj_w  (f32)
  gemm_kernel<0, 1><<<dim3(HDIM / 64, S_LEN / 128, 1), 256, 0, stream>>>(
      ob, HDIM, 0, o_proj_w, HDIM, 0, hbuf, HDIM, 0, hidden, HDIM, HDIM);
  // 11) x2 = rms(h, ln2_w) -> bf16
  rmsnorm_bf16_kernel<<<S_LEN, 256, 0, stream>>>(hbuf, ln2_w, x2b, HDIM);
  // 12) act = silu(x2 @ Wg) * (x2 @ Wu)  -> bf16 (fused, no f32 gate_up buf)
  gateup_kernel<<<dim3(IDIM / 64, S_LEN / 128, 1), 256, 0, stream>>>(
      x2b, HDIM, gate_up_w, act);
  // 13) out = h + act @ down_w  (f32, final residual)
  gemm_kernel<0, 1><<<dim3(HDIM / 64, S_LEN / 128, 1), 256, 0, stream>>>(
      act, IDIM, 0, down_w, HDIM, 0, out, HDIM, 0, hbuf, HDIM, IDIM);
}
